// AnchorGraphLearner_62715112456876
// MI455X (gfx1250) — matrix operand order repaired and checked by reading, not compiled, not run on hardware
//
#include <hip/hip_runtime.h>

#define NCTX   8192
#define FDIM   256
#define ANCH   256
#define HEADS  16
#define TILE_N 32

typedef __attribute__((ext_vector_type(16))) _Float16 v16h;
typedef __attribute__((ext_vector_type(8)))  float    v8f;
typedef __attribute__((ext_vector_type(2)))  _Float16 h2v;

union ABFrag { v16h h; unsigned int u[8]; };
union F4     { float4 v; float f[4]; };
union U4     { uint4 v; unsigned int u[4]; };
union H2U    { h2v h; unsigned int u; };

__device__ __forceinline__ unsigned int pack2h(float a, float b) {
  union { _Float16 h[2]; unsigned int u; } p;
  p.h[0] = (_Float16)a; p.h[1] = (_Float16)b;
  return p.u;
}

// Stage 32 rows x 256 cols of f32 into packed-f16 LDS x_s[32][132 pairs].
__device__ __forceinline__ void stage_rows_f16(unsigned int* x_s, const float* src,
                                               int row0, int tid) {
  const int r  = tid >> 3;
  const int c0 = (tid & 7) * 32;
  const float* p = src + (size_t)(row0 + r) * FDIM + c0;
#pragma unroll
  for (int i = 0; i < 8; ++i) {
    float4 v = *(const float4*)(p + 4 * i);
    x_s[r * 132 + (c0 >> 1) + 2 * i]     = pack2h(v.x, v.y);
    x_s[r * 132 + (c0 >> 1) + 2 * i + 1] = pack2h(v.z, v.w);
  }
}

// hx_s[f][n] = sum_k X[n][k] * W[f*512 + k], f in [0,256), n in [0,32).
__device__ __forceinline__ void tile_gemm_f16(const unsigned int* x_s,
                                              unsigned int* w_s, float* hx_s,
                                              const float* Wrow, int tid) {
  const int lane = tid & 31;
  const int wv   = tid >> 5;     // wave 0..7 -> f-tiles {wv, wv+8}
  const int l15  = lane & 15;
  const int lhi  = lane >> 4;

  v8f c[2][2];
#pragma unroll
  for (int a = 0; a < 2; ++a)
#pragma unroll
    for (int b = 0; b < 2; ++b)
#pragma unroll
      for (int e = 0; e < 8; ++e) c[a][b][e] = 0.0f;

#pragma unroll
  for (int kc64 = 0; kc64 < 4; ++kc64) {
    __syncthreads();
    {  // stage 64-wide K chunk of W: row f = tid
      const float* p = Wrow + (size_t)tid * 512 + kc64 * 64;
#pragma unroll
      for (int i = 0; i < 16; ++i) {
        float4 v = *(const float4*)(p + 4 * i);
        w_s[tid * 33 + 2 * i]     = pack2h(v.x, v.y);
        w_s[tid * 33 + 2 * i + 1] = pack2h(v.z, v.w);
      }
      if (kc64 < 3) {  // global_prefetch_b8: next K chunk of this W row
        __builtin_prefetch(p + 64, 0, 1);
        __builtin_prefetch(p + 96, 0, 1);
      }
    }
    __syncthreads();
#pragma unroll
    for (int sub = 0; sub < 2; ++sub) {
      const int kp0 = kc64 * 32 + sub * 16;   // A pair base (global K/2)
      ABFrag afr[2], bfr[2];
#pragma unroll
      for (int nt = 0; nt < 2; ++nt) {        // A: 16x32 f16, ISA layout
        const int row  = nt * 16 + l15;
        const int base = kp0 + (lhi ? 4 : 0);
#pragma unroll
        for (int v = 0; v < 8; ++v)
          afr[nt].u[v] = x_s[row * 132 + base + (v < 4 ? v : v + 4)];
      }
#pragma unroll
      for (int fi = 0; fi < 2; ++fi) {        // B: 32x16 f16, col per lane
        const int frow = (wv + fi * 8) * 16 + l15;
        const int pb   = sub * 16 + lhi * 4;
#pragma unroll
        for (int v = 0; v < 8; ++v)
          bfr[fi].u[v] = w_s[frow * 33 + pb + (v < 4 ? v : v + 4)];
      }
#pragma unroll
      for (int nt = 0; nt < 2; ++nt)
#pragma unroll
        for (int fi = 0; fi < 2; ++fi)
          c[nt][fi] = __builtin_amdgcn_wmma_f32_16x16x32_f16(
              false, afr[nt].h, false, bfr[fi].h, (short)0, c[nt][fi],
              false, false);
    }
  }
  __syncthreads();
#pragma unroll
  for (int nt = 0; nt < 2; ++nt)
#pragma unroll
    for (int fi = 0; fi < 2; ++fi) {          // C layout: VGPR j -> M=j(+8)
      const int fr = (wv + fi * 8) * 16 + l15;
#pragma unroll
      for (int j = 0; j < 8; ++j)
        hx_s[fr * 36 + nt * 16 + lhi * 8 + j] = c[nt][fi][j];
    }
  __syncthreads();
}

__global__ void copy_anchors_kernel(const float* __restrict__ anchors,
                                    float* __restrict__ dst) {
  int i = blockIdx.x * blockDim.x + threadIdx.x;   // float4 index
  ((float4*)dst)[i] = ((const float4*)anchors)[i];
}

// ha_ws[h][fpair][a] = packed f16 pair over f of (anchors@W1a^T + b1).
__global__ void ha_precompute_kernel(const float* __restrict__ anchors,
                                     const float* __restrict__ W1,
                                     const float* __restrict__ b1,
                                     unsigned int* __restrict__ ha_ws) {
  __shared__ unsigned int x_s[32 * 132];
  __shared__ unsigned int w_s[256 * 33];
  __shared__ float        hx_s[256 * 36];
  const int tid = threadIdx.x;
  const int at  = blockIdx.x;   // anchor tile 0..7
  const int h   = blockIdx.y;   // head

  stage_rows_f16(x_s, anchors, at * 32, tid);
  tile_gemm_f16(x_s, w_s, hx_s,
                W1 + (size_t)(h * FDIM) * 512 + FDIM, tid);   // W1a part

  // pack pairs along f; thread -> fpair = tid&127, anchor half = tid>>7
  const int fp = tid & 127;
  const int ah = tid >> 7;            // 0/1 -> 16 anchors each
  const float bb0 = b1[h * FDIM + 2 * fp];
  const float bb1 = b1[h * FDIM + 2 * fp + 1];
  unsigned int* dst =
      ha_ws + ((size_t)(h * 128 + fp) * ANCH) + at * 32 + ah * 16;
#pragma unroll
  for (int q4 = 0; q4 < 4; ++q4) {
    U4 v;
#pragma unroll
    for (int e = 0; e < 4; ++e) {
      const int col = ah * 16 + q4 * 4 + e;
      v.u[e] = pack2h(hx_s[(2 * fp) * 36 + col] + bb0,
                      hx_s[(2 * fp + 1) * 36 + col] + bb1);
    }
    *(uint4*)(dst + q4 * 4) = v.v;
  }
}

__global__ void __launch_bounds__(256, 2)
fused_attn_kernel(const float* __restrict__ ctx, const float* __restrict__ W1,
                  const float* __restrict__ W2, const float* __restrict__ b2,
                  const unsigned int* __restrict__ ha_ws,
                  float* __restrict__ out) {
  __shared__ unsigned int x_s[32 * 132];
  __shared__ unsigned int w_s[256 * 33];
  __shared__ float        hx_s[256 * 36];   // [f][n] f32
  __shared__ unsigned int ha_s[32 * 264];   // [fpair_local][a] packed f16
  __shared__ unsigned int w2p_s[128];       // packed f16 pairs of w2

  const int tid = threadIdx.x;
  const int nb  = blockIdx.x * TILE_N;
  const int ng  = tid >> 5;   // 4 context rows each
  const int ag  = tid & 31;   // 8 anchors each
  const int fl  = tid >> 3;        // staging fpair row
  const int a0  = (tid & 7) * 32;  // staging anchor base

  stage_rows_f16(x_s, ctx, nb, tid);
  // warm up L2/WGP$ for head 0 ha chunk 0
  __builtin_prefetch(ha_ws + (size_t)fl * ANCH + a0, 0, 1);

  h2v hzero;
  hzero[0] = (_Float16)0.0f; hzero[1] = (_Float16)0.0f;

  float facc[4][8];
#pragma unroll
  for (int i = 0; i < 4; ++i)
#pragma unroll
    for (int j = 0; j < 8; ++j) facc[i][j] = 0.0f;

  for (int h = 0; h < HEADS; ++h) {
    tile_gemm_f16(x_s, w_s, hx_s, W1 + (size_t)(h * FDIM) * 512, tid); // W1x
    if (tid < 128)
      w2p_s[tid] = pack2h(W2[h * FDIM + 2 * tid], W2[h * FDIM + 2 * tid + 1]);
    const float b2h = b2[h];

    h2v s2[4][8];
#pragma unroll
    for (int i = 0; i < 4; ++i)
#pragma unroll
      for (int j = 0; j < 8; ++j) s2[i][j] = hzero;

    for (int fc = 0; fc < 4; ++fc) {     // 32 fpairs (64 f) per chunk
      __syncthreads();
      {  // stage packed ha chunk [32 fpair][256 a], coalesced
        const unsigned int* p =
            ha_ws + ((size_t)(h * 128 + fc * 32 + fl) * ANCH) + a0;
#pragma unroll
        for (int i = 0; i < 8; ++i)
          *(uint4*)&ha_s[fl * 264 + a0 + 4 * i] = *(const uint4*)(p + 4 * i);
        // prefetch next chunk (or next head's first chunk) into caches
        const int nh = (fc < 3) ? h : h + 1;
        const int nc = (fc < 3) ? fc + 1 : 0;
        if (nh < HEADS)
          __builtin_prefetch(
              ha_ws + ((size_t)(nh * 128 + nc * 32 + fl) * ANCH) + a0, 0, 1);
      }
      __syncthreads();
#pragma unroll 2
      for (int fp = 0; fp < 32; ++fp) {
        const int f0 = fc * 64 + fp * 2;
        H2U w2u; w2u.u = w2p_s[fc * 32 + fp];
        F4 hxa, hxb;
        hxa.v = *(const float4*)&hx_s[f0 * 36 + ng * 4];
        hxb.v = *(const float4*)&hx_s[(f0 + 1) * 36 + ng * 4];
        H2U hxp[4];
#pragma unroll
        for (int i = 0; i < 4; ++i) hxp[i].u = pack2h(hxa.f[i], hxb.f[i]);
        U4 haA, haB;
        haA.v = *(const uint4*)&ha_s[fp * 264 + ag * 8];
        haB.v = *(const uint4*)&ha_s[fp * 264 + ag * 8 + 4];
#pragma unroll
        for (int i = 0; i < 4; ++i)
#pragma unroll
          for (int j = 0; j < 4; ++j) {
            H2U ha0; ha0.u = haA.u[j];
            h2v t0 = hxp[i].h + ha0.h;                    // v_pk_add_f16
            t0 = __builtin_elementwise_max(t0, hzero);    // v_pk_max_f16
            s2[i][j] = __builtin_elementwise_fma(t0, w2u.h, s2[i][j]);
            H2U ha1; ha1.u = haB.u[j];
            h2v t1 = hxp[i].h + ha1.h;
            t1 = __builtin_elementwise_max(t1, hzero);
            s2[i][j + 4] = __builtin_elementwise_fma(t1, w2u.h, s2[i][j + 4]);
          }
      }
    }
#pragma unroll
    for (int i = 0; i < 4; ++i)
#pragma unroll
      for (int j = 0; j < 8; ++j) {
        const float sv = (float)s2[i][j][0] + (float)s2[i][j][1];
        facc[i][j] += 1.0f / (1.0f + __expf(-(sv + b2h)));
      }
    __syncthreads();
  }

  const float inv = 1.0f / (float)HEADS;
#pragma unroll
  for (int i = 0; i < 4; ++i) {
    float* dst = out + (size_t)(nb + ng * 4 + i) * ANCH + ag * 8;
    float4 v0, v1;
    v0.x = facc[i][0] * inv; v0.y = facc[i][1] * inv;
    v0.z = facc[i][2] * inv; v0.w = facc[i][3] * inv;
    v1.x = facc[i][4] * inv; v1.y = facc[i][5] * inv;
    v1.z = facc[i][6] * inv; v1.w = facc[i][7] * inv;
    *(float4*)dst       = v0;
    *(float4*)(dst + 4) = v1;
  }
}

extern "C" void kernel_launch(void* const* d_in, const int* in_sizes, int n_in,
                              void* d_out, int out_size, void* d_ws,
                              size_t ws_size, hipStream_t stream) {
  const float* ctx     = (const float*)d_in[0];
  const float* anchors = (const float*)d_in[1];
  const float* W1      = (const float*)d_in[2];
  const float* b1      = (const float*)d_in[3];
  const float* W2      = (const float*)d_in[4];
  const float* b2      = (const float*)d_in[5];
  float* out = (float*)d_out;
  unsigned int* ha_ws = (unsigned int*)d_ws;  // HEADS*128*ANCH u32 = 2 MB

  copy_anchors_kernel<<<(ANCH * FDIM) / (4 * 256), 256, 0, stream>>>(
      anchors, out + (size_t)NCTX * ANCH);
  ha_precompute_kernel<<<dim3(ANCH / 32, HEADS), 256, 0, stream>>>(
      anchors, W1, b1, ha_ws);
  fused_attn_kernel<<<NCTX / TILE_N, 256, 0, stream>>>(
      ctx, W1, W2, b2, ha_ws, out);
}
